// Image_Encoder_16243566313486
// MI455X (gfx1250) — matrix operand order repaired and checked by reading, not compiled, not run on hardware
//
#include <hip/hip_runtime.h>
#include <hip/hip_bf16.h>

typedef _Float16 v16h __attribute__((ext_vector_type(16)));
typedef float    v8f  __attribute__((ext_vector_type(8)));

// Reduction-index mapping (shared by weight pack + conv):
//   K = (ci*5 + kh)*8 + kw   with kw padded 5 -> 8 (kw in [0,8), valid kw < 5)
//   Kpad = Cin*40  (always a multiple of 32 since Cin % 8 == 0)
// kw is a compile-time constant in the unrolled gather; only two div-by-5
// (by-constant -> mul/shift) per K-tile.

// ---------------------------------------------------------------------------
// 1) concat(img_occ, img_unocc) on channel dim + f32 -> f16
// ---------------------------------------------------------------------------
__global__ void concat_f16_kernel(const float* __restrict__ occ,
                                  const float* __restrict__ unocc,
                                  _Float16* __restrict__ x0) {
    const size_t PIX = 256u * 256u;
    const size_t total = (size_t)16 * 8 * PIX;
    size_t i = (size_t)blockIdx.x * blockDim.x + threadIdx.x;
    if (i >= total) return;
    size_t b = i / (8 * PIX);
    size_t r = i % (8 * PIX);
    size_t c = r / PIX;
    size_t p = r % PIX;
    float v = (c < 4) ? occ[(b * 4 + c) * PIX + p]
                      : unocc[(b * 4 + (c - 4)) * PIX + p];
    x0[i] = (_Float16)v;
}

// ---------------------------------------------------------------------------
// 2) Pack conv weights into the wave32 WMMA A-fragment layout (16x32 f16).
//    Buffer layout: [mtile][ktile][lane(32)][elem(16)] contiguous f16.
//    Contiguous K order, matching the B gather (16-bit A 16x32): lane(0-15)=M rows; elem j -> VGPR j/2,
//    half j&1; K-in-tile = laneHalf*16 + grp*8 + wi*2 + h.
// ---------------------------------------------------------------------------
__global__ void pack_weights_kernel(const float* __restrict__ W,
                                    _Float16* __restrict__ pw,
                                    int cin, int cout, int kpad) {
    int tid = blockIdx.x * blockDim.x + threadIdx.x;
    int total = cout * kpad;          // == mtiles * nkt * 512
    if (tid >= total) return;
    int nkt  = kpad >> 5;
    int ktm  = tid >> 9;              // mtile*nkt + kt
    int rem  = tid & 511;
    int lane = rem >> 4;
    int j    = rem & 15;
    int mtile = ktm / nkt;
    int kt    = ktm - mtile * nkt;
    int M    = lane & 15;
    int half = lane >> 4;
    int v    = j >> 1, h = j & 1;
    int grp  = v >> 2, wi = v & 3;
    int K    = kt * 32 + half * 16 + grp * 8 + wi * 2 + h;
    int co   = mtile * 16 + M;
    int kw   = K & 7;                 // padded tap column
    int t    = K >> 3;                // ci*5 + kh  (always < cin*5)
    float val = 0.f;
    if (kw < 5) {
        int ci = t / 5;
        int kh = t - ci * 5;
        val = W[(((size_t)co * cin + ci) * 5 + kh) * 5 + kw];
    }
    pw[tid] = (_Float16)val;
}

// ---------------------------------------------------------------------------
// 3) Fuse conv bias + BatchNorm into per-channel scale/bias:
//    y = conv*inv + ((b - mean)*inv + beta),  inv = gamma/sqrt(var+eps)
// ---------------------------------------------------------------------------
__global__ void make_sb_kernel(const float* __restrict__ b,
                               const float* __restrict__ g,
                               const float* __restrict__ beta,
                               const float* __restrict__ mean,
                               const float* __restrict__ var,
                               float* __restrict__ sb, int cout) {
    int i = blockIdx.x * blockDim.x + threadIdx.x;
    if (i >= cout) return;
    float inv = g[i] * rsqrtf(var[i] + 1e-5f);
    sb[2 * i]     = inv;
    sb[2 * i + 1] = (b[i] - mean[i]) * inv + beta[i];
}

// ---------------------------------------------------------------------------
// K-loop body for the implicit-GEMM conv. FAST=true: tile is fully interior
// (wave-uniform condition) -> no per-element predication at all.
// ---------------------------------------------------------------------------
template <int MT, bool FAST>
__device__ __forceinline__ void conv_kloop(
    int nkt, int half, int iy0, int ix0, int Hin, int Win, bool nvalid,
    const _Float16* __restrict__ xb, const _Float16* const (&aw)[MT],
    v8f (&acc)[MT]) {
    for (int kt = 0; kt < nkt; ++kt) {
        // B fragment: lanes 0-15 hold K = kt*32 + j, lanes 16-31 the next 16.
        // Each lane-half covers exactly two (ci,kh) rows; kw == j is constant.
        union { v16h v; _Float16 h[16]; } bu;
        const int t0 = kt * 4 + half * 2;  // = (kt*32 + half*16) >> 3
        int ci0 = t0 / 5;
        int kh0 = t0 - ci0 * 5;
        int ci1 = (t0 + 1) / 5;
        int kh1 = (t0 + 1) - ci1 * 5;
        const int iyA = iy0 + kh0;
        const int iyB = iy0 + kh1;
        const _Float16* rowA = xb + ((size_t)ci0 * Hin + iyA) * Win + ix0;
        const _Float16* rowB = xb + ((size_t)ci1 * Hin + iyB) * Win + ix0;
        bool okA = true, okB = true;
        if (!FAST) {
            okA = nvalid && ((unsigned)iyA < (unsigned)Hin);
            okB = nvalid && ((unsigned)iyB < (unsigned)Hin);
        }
#pragma unroll
        for (int j = 0; j < 8; ++j) {
            _Float16 vA = (_Float16)0.f;
            _Float16 vB = (_Float16)0.f;
            if (j < 5) {                            // kw >= 5 is padding
                if (FAST) {
                    vA = rowA[j];                   // unconditional imm-offset
                    vB = rowB[j];                   // d16 loads
                } else {
                    const bool inx = (unsigned)(ix0 + j) < (unsigned)Win;
                    if (okA && inx) vA = rowA[j];
                    if (okB && inx) vB = rowB[j];
                }
            }
            bu.h[j]     = vA;
            bu.h[j + 8] = vB;
        }
        // MT independent WMMAs share this B fragment (EXEC all-1s here).
#pragma unroll
        for (int m = 0; m < MT; ++m) {
            v16h a = *(const v16h*)(aw[m] + (size_t)kt * 512);
            acc[m] = __builtin_amdgcn_wmma_f32_16x16x32_f16(
                false, a, false, bu.v, (short)0, acc[m], false, false);
        }
    }
}

// ---------------------------------------------------------------------------
// 4) Implicit-GEMM conv + BN + ReLU via V_WMMA_F32_16X16X32_F16.
//    One wave32 computes a (MT*16)(Cout) x 16(pixels) tile: one im2col B
//    gather feeds MT independent WMMAs. 4 independent waves per block.
// ---------------------------------------------------------------------------
template <int MT>
__global__ void __launch_bounds__(128)
conv_wmma_kernel(const _Float16* __restrict__ xin,
                 const _Float16* __restrict__ pw,
                 const float* __restrict__ sb,
                 _Float16* __restrict__ xout,
                 int Cin, int Cout, int Hin, int Win,
                 int Hout, int Wout, int stride,
                 int nkt, int ntiles) {
    const int wave = threadIdx.x >> 5;
    const int lane = threadIdx.x & 31;
    const int ntile = blockIdx.x * 4 + wave;
    if (ntile >= ntiles) return;           // wave-uniform exit
    const int mtile0 = blockIdx.y * MT;
    const int bidx   = blockIdx.z;
    const int Npix   = Hout * Wout;
    const int half   = lane >> 4;          // 0 or 1 (K half / M half)
    const int nloc   = lane & 15;          // N column within tile
    const int n      = ntile * 16 + nloc;
    const bool nvalid = (n < Npix);
    int oy = 0, ox = 0;
    if (nvalid) { oy = n / Wout; ox = n - oy * Wout; }
    const int iy0 = oy * stride - 1;       // padding = 1
    const int ix0 = ox * stride - 1;

    // Wave-uniform interior test (scalar inputs only): whole tile in one
    // output row and every 5x5 tap of every lane lands in-bounds.
    const int n0  = ntile * 16;
    const int oy0 = n0 / Wout;
    const int ox0 = n0 - oy0 * Wout;
    const bool fast =
        (n0 + 15 < Npix) && (ox0 + 15 < Wout) &&
        (ox0 * stride - 1 >= 0) && ((ox0 + 15) * stride - 1 + 4 < Win) &&
        (oy0 * stride - 1 >= 0) && (oy0 * stride - 1 + 4 < Hin);

    const _Float16* __restrict__ xb = xin + (size_t)bidx * Cin * Hin * Win;
    const _Float16* aw[MT];
#pragma unroll
    for (int m = 0; m < MT; ++m)
        aw[m] = pw + (size_t)(mtile0 + m) * nkt * 512 + (size_t)lane * 16;

    v8f acc[MT];
#pragma unroll
    for (int m = 0; m < MT; ++m) acc[m] = (v8f){};

    if (fast)
        conv_kloop<MT, true>(nkt, half, iy0, ix0, Hin, Win, nvalid, xb, aw, acc);
    else
        conv_kloop<MT, false>(nkt, half, iy0, ix0, Hin, Win, nvalid, xb, aw, acc);

    if (!nvalid) return;
    const size_t obase = (size_t)bidx * Cout * Npix + (size_t)oy * Wout + ox;
#pragma unroll
    for (int m = 0; m < MT; ++m) {
#pragma unroll
        for (int v = 0; v < 8; ++v) {
            int co = (mtile0 + m) * 16 + v + half * 8;  // lanes16-31 -> M+8
            float s = sb[2 * co];
            float t = sb[2 * co + 1];
            float val = acc[m][v] * s + t;
            val = val > 0.f ? val : 0.f;                // ReLU
            xout[obase + (size_t)co * Npix] = (_Float16)val;
        }
    }
}

// ---------------------------------------------------------------------------
// 5) Vertex projection + bilinear pooling over the 4 selected feature maps.
//    Reproduces the reference floor/ceil quirk (integer coord -> zero weight).
// ---------------------------------------------------------------------------
__global__ void __launch_bounds__(128)
pool_kernel(const _Float16* __restrict__ f0, const _Float16* __restrict__ f1,
            const _Float16* __restrict__ f2, const _Float16* __restrict__ f3,
            const float* __restrict__ verts, const float* __restrict__ cam,
            float* __restrict__ out) {
    const int N = 8192;
    int vid = blockIdx.x;                 // b*N + n
    int b = vid / N;
    const float* v = verts + (size_t)vid * 3;
    float vx = v[0], vy = v[1], vz = v[2];
    float e0 = cam[0] * vx + cam[1] * vy + cam[2]  * vz + cam[3];
    float e1 = cam[4] * vx + cam[5] * vy + cam[6]  * vz + cam[7];
    float e2 = cam[8] * vx + cam[9] * vy + cam[10] * vz + cam[11];
    float xs = e1 / e2 / 256.f;
    float ys = e0 / e2 / 256.f;

    const _Float16* fptr[4] = {f0, f1, f2, f3};
    const int Cs[4]   = {32, 64, 128, 256};
    const int Ds[4]   = {57, 24, 7, 3};
    const int Coff[4] = {0, 32, 96, 224};
    float* ob = out + (size_t)vid * 480;

    for (int blk = 0; blk < 4; ++blk) {
        const int C = Cs[blk], dim = Ds[blk];
        float cx = fminf(fmaxf(xs * (float)dim, 0.f), (float)(dim - 1));
        float cy = fminf(fmaxf(ys * (float)dim, 0.f), (float)(dim - 1));
        float x1 = floorf(cx), x2 = ceilf(cx);
        float y1 = floorf(cy), y2 = ceilf(cy);
        float A  = x2 - cx, Bw = cx - x1;
        float G  = y2 - cy, H  = cy - y1;
        int x1i = (int)x1, x2i = (int)x2, y1i = (int)y1, y2i = (int)y2;
        int i11 = x1i * dim + y1i, i12 = x1i * dim + y2i;
        int i21 = x2i * dim + y1i, i22 = x2i * dim + y2i;
        float wA = A * G, wB = A * H, wC = Bw * G, wD = Bw * H;
        const _Float16* fb = fptr[blk] + (size_t)b * C * dim * dim;
        for (int c = threadIdx.x; c < C; c += blockDim.x) {
            const _Float16* p = fb + (size_t)c * dim * dim;
            float Cg = (float)p[i11];
            float Dg = (float)p[i12];
            float Eg = (float)p[i21];
            float Fg = (float)p[i22];
            ob[Coff[blk] + c] = wA * Cg + wB * Dg + wC * Eg + wD * Fg;
        }
    }
}

// ---------------------------------------------------------------------------
// Host side
// ---------------------------------------------------------------------------
extern "C" void kernel_launch(void* const* d_in, const int* in_sizes, int n_in,
                              void* d_out, int out_size, void* d_ws, size_t ws_size,
                              hipStream_t stream) {
    (void)in_sizes; (void)n_in; (void)out_size;
    const int B = 16;
    static const int CIN[13]  = {8, 16, 16, 16, 32, 32, 32, 64, 64, 64, 128, 128, 128};
    static const int COUT[13] = {16, 16, 16, 32, 32, 32, 64, 64, 64, 128, 128, 128, 256};
    static const int DIN[13]  = {256, 127, 125, 123, 61, 59, 57, 28, 26, 24, 11, 9, 7};
    static const int DOUT[13] = {127, 125, 123, 61, 59, 57, 28, 26, 24, 11, 9, 7, 3};

    const float* img_occ   = (const float*)d_in[0];
    const float* img_unocc = (const float*)d_in[1];
    const float* verts     = (const float*)d_in[2];
    const float* camera    = (const float*)d_in[3];

    char* ws = (char*)d_ws;
    size_t off = 0;
    auto alloc = [&](size_t bytes) -> size_t {
        size_t o = off;
        off = (off + bytes + 255) & ~(size_t)255;
        return o;
    };

    size_t x0off = alloc((size_t)B * 8 * 256 * 256 * sizeof(_Float16));
    size_t actoff[13], pwoff[13], sboff[13];
    int kpad[13];
    for (int e = 0; e < 13; ++e) {
        actoff[e] = alloc((size_t)B * COUT[e] * DOUT[e] * DOUT[e] * sizeof(_Float16));
        kpad[e] = CIN[e] * 40;  // (ci*5+kh) rows x 8 padded kw; multiple of 32
        pwoff[e] = alloc((size_t)COUT[e] * kpad[e] * sizeof(_Float16));
        sboff[e] = alloc((size_t)2 * COUT[e] * sizeof(float));
    }
    if (off > ws_size) return;  // workspace too small — nothing safe to do

    _Float16* x0 = (_Float16*)(ws + x0off);

    // 1) build f16 8-channel input
    {
        size_t total = (size_t)B * 8 * 256 * 256;
        int blocks = (int)((total + 255) / 256);
        concat_f16_kernel<<<blocks, 256, 0, stream>>>(img_occ, img_unocc, x0);
    }

    // 2/3) pack weights + fused scale/bias for all executed layers
    for (int e = 0; e < 13; ++e) {
        const float* W    = (const float*)d_in[4 + 6 * e + 0];
        const float* bia  = (const float*)d_in[4 + 6 * e + 1];
        const float* gam  = (const float*)d_in[4 + 6 * e + 2];
        const float* bet  = (const float*)d_in[4 + 6 * e + 3];
        const float* mea  = (const float*)d_in[4 + 6 * e + 4];
        const float* var  = (const float*)d_in[4 + 6 * e + 5];
        int total = COUT[e] * kpad[e];
        pack_weights_kernel<<<(total + 255) / 256, 256, 0, stream>>>(
            W, (_Float16*)(ws + pwoff[e]), CIN[e], COUT[e], kpad[e]);
        make_sb_kernel<<<1, 256, 0, stream>>>(
            bia, gam, bet, mea, var, (float*)(ws + sboff[e]), COUT[e]);
    }

    // 4) conv stack (stream order serializes dependencies)
    for (int e = 0; e < 13; ++e) {
        int hout = DOUT[e];
        int npix = hout * hout;
        int ntiles = (npix + 15) / 16;
        int mt = (COUT[e] % 64 == 0) ? 4 : ((COUT[e] % 32 == 0) ? 2 : 1);
        dim3 grid((ntiles + 3) / 4, COUT[e] / (16 * mt), B);
        const _Float16* xinp = (e == 0) ? x0 : (const _Float16*)(ws + actoff[e - 1]);
        const _Float16* pwp = (const _Float16*)(ws + pwoff[e]);
        const float*    sbp = (const float*)(ws + sboff[e]);
        _Float16*       xop = (_Float16*)(ws + actoff[e]);
        int stride = (e % 3 == 0) ? 2 : 1;
        int nkt = kpad[e] / 32;
        switch (mt) {
        case 4:
            conv_wmma_kernel<4><<<grid, dim3(128), 0, stream>>>(
                xinp, pwp, sbp, xop, CIN[e], COUT[e], DIN[e], DIN[e],
                hout, hout, stride, nkt, ntiles);
            break;
        case 2:
            conv_wmma_kernel<2><<<grid, dim3(128), 0, stream>>>(
                xinp, pwp, sbp, xop, CIN[e], COUT[e], DIN[e], DIN[e],
                hout, hout, stride, nkt, ntiles);
            break;
        default:
            conv_wmma_kernel<1><<<grid, dim3(128), 0, stream>>>(
                xinp, pwp, sbp, xop, CIN[e], COUT[e], DIN[e], DIN[e],
                hout, hout, stride, nkt, ntiles);
            break;
        }
    }

    // 5) pooling over feature maps e=5, 8, 11, 12 -> [16, 8192, 480] f32
    pool_kernel<<<B * 8192, 128, 0, stream>>>(
        (const _Float16*)(ws + actoff[5]),
        (const _Float16*)(ws + actoff[8]),
        (const _Float16*)(ws + actoff[11]),
        (const _Float16*)(ws + actoff[12]),
        verts, camera, (float*)d_out);
}
